// CAM_Module_10479720202888
// MI455X (gfx1250) — matrix operand order repaired and checked by reading, not compiled, not run on hardware
//
#include <hip/hip_runtime.h>

// ---------------------------------------------------------------------------
// CAM module (channel attention) for MI455X / gfx1250, wave32 + WMMA bf16.
// B=16, C=512, N=H*W=4096.
// ---------------------------------------------------------------------------

typedef __attribute__((ext_vector_type(16))) __bf16 v16bf;
typedef __attribute__((ext_vector_type(8)))  float  v8f;

union Frag16 {
    v16bf        v;
    unsigned int d[8];
    uint4        q[2];
};

#define CAM_B 16
#define CAM_C 512
#define CAM_N 4096

__device__ __forceinline__ unsigned short f2bf(float f) {
    unsigned int u = __builtin_bit_cast(unsigned int, f);
    u += 0x7fffu + ((u >> 16) & 1u);   // round-to-nearest-even
    return (unsigned short)(u >> 16);
}

// ---------------------------------------------------------------------------
// Kernel 0: x (fp32, B*C*N) -> xb (bf16, same layout) and xbT (bf16, [b][n][c])
// 64x64 LDS-tiled transpose, coalesced reads and writes.
// grid = B * (C/64) * (N/64) = 16*8*64 = 8192, block = 256
// ---------------------------------------------------------------------------
__global__ __launch_bounds__(256) void k0_convert(
    const float* __restrict__ x,
    unsigned short* __restrict__ xb,
    unsigned short* __restrict__ xbT)
{
    __shared__ unsigned short tile[64 * 65];
    const int bid = blockIdx.x;
    const int b   = bid >> 9;            // / (8*64)
    const int rem = bid & 511;
    const int c0  = (rem >> 6) << 6;     // channel tile base
    const int n0  = (rem & 63) << 6;     // spatial tile base
    const int t   = threadIdx.x;

#pragma unroll
    for (int i = 0; i < 16; ++i) {
        const int lin = i * 256 + t;
        const int r  = lin >> 6;
        const int cc = lin & 63;
        const size_t idx = ((size_t)(b * CAM_C + c0 + r)) * CAM_N + (n0 + cc);
        const unsigned short v = f2bf(x[idx]);
        xb[idx] = v;
        tile[r * 65 + cc] = v;
    }
    __syncthreads();
#pragma unroll
    for (int i = 0; i < 16; ++i) {
        const int lin = i * 256 + t;
        const int rn = lin >> 6;
        const int cc = lin & 63;
        xbT[((size_t)(b * CAM_N + n0 + rn)) * CAM_C + (c0 + cc)] = tile[cc * 65 + rn];
    }
}

// ---------------------------------------------------------------------------
// Kernel 1: energy[b] = v * v^T (C x C over N), WMMA bf16, symmetric.
// WG = 256 threads (8 waves) computes one 128x128 tile; only upper-triangular
// block pairs are computed, mirrored on store. Each wave: 2x4 sub-tiles of
// 16x16 => 8 WMMAs per K-step. LDS double-buffered, next K-step prefetched
// into registers so global latency overlaps compute.
// grid = B * 10 = 160, block = 256
// ---------------------------------------------------------------------------
__global__ __launch_bounds__(256) void k1_gram(
    const unsigned short* __restrict__ xb,
    float* __restrict__ energy)
{
    __shared__ unsigned int lds[8192];   // 2 buffers x (A 2048 + B 2048) dwords

    static const int pi[10] = {0,0,0,0,1,1,1,2,2,3};
    static const int pj[10] = {0,1,2,3,1,2,3,2,3,3};

    const int b   = blockIdx.x / 10;
    const int k10 = blockIdx.x - b * 10;
    const int i0  = pi[k10] * 128;
    const int j0  = pj[k10] * 128;

    const int t    = threadIdx.x;
    const int lane = t & 31;
    const int wid  = t >> 5;
    const int ti   = (wid >> 1) << 5;    // wave row base within 128 (0,32,64,96)
    const int tj   = (wid & 1) << 6;     // wave col base within 128 (0,64)
    const bool hi  = lane >= 16;
    const int  l   = lane & 15;

    v8f acc[2][4];
#pragma unroll
    for (int g = 0; g < 2; ++g)
#pragma unroll
        for (int c = 0; c < 4; ++c)
#pragma unroll
            for (int i = 0; i < 8; ++i) acc[g][c][i] = 0.0f;

    // stage-load helper data: each thread loads 4 uint4 per K-step
    // idx = q*256 + t ; idx<512 -> A panel, else B panel
    uint4 pre[4];
#pragma unroll
    for (int q = 0; q < 4; ++q) {
        const int idx = q * 256 + t;
        const int isB = idx >> 9;
        const int jx  = idx & 511;
        const int r   = jx >> 2;
        const int seg = jx & 3;
        const int row = (isB ? j0 : i0) + r;
        pre[q] = *(const uint4*)(xb + ((size_t)(b * CAM_C + row)) * CAM_N + seg * 8);
    }

    int p = 0;
    for (int kk = 0; kk < CAM_N / 32; ++kk) {
        __syncthreads();
#pragma unroll
        for (int q = 0; q < 4; ++q)
            ((uint4*)lds)[p * 1024 + q * 256 + t] = pre[q];
        __syncthreads();

        if (kk < CAM_N / 32 - 1) {
            const int k0 = (kk + 1) * 32;
#pragma unroll
            for (int q = 0; q < 4; ++q) {
                const int idx = q * 256 + t;
                const int isB = idx >> 9;
                const int jx  = idx & 511;
                const int r   = jx >> 2;
                const int seg = jx & 3;
                const int row = (isB ? j0 : i0) + r;
                pre[q] = *(const uint4*)(xb + ((size_t)(b * CAM_C + row)) * CAM_N + k0 + seg * 8);
            }
        }

        const int base = p * 4096;
        Frag16 a[2], bf[4];
#pragma unroll
        for (int g = 0; g < 2; ++g) {
            const int ab = base + (ti + g * 16 + l) * 16 + (hi ? 4 : 0);
            a[g].q[0] = *(const uint4*)&lds[ab];
            a[g].q[1] = *(const uint4*)&lds[ab + 8];
        }
#pragma unroll
        for (int c = 0; c < 4; ++c) {
            const int bb = base + 2048 + (tj + c * 16 + l) * 16 + (hi ? 4 : 0);
            bf[c].q[0] = *(const uint4*)&lds[bb];
            bf[c].q[1] = *(const uint4*)&lds[bb + 8];
        }
#pragma unroll
        for (int g = 0; g < 2; ++g)
#pragma unroll
            for (int c = 0; c < 4; ++c)
                acc[g][c] = __builtin_amdgcn_wmma_f32_16x16x32_bf16(
                    false, a[g].v, false, bf[c].v, (short)0, acc[g][c], false, false);
        p ^= 1;
    }

#pragma unroll
    for (int g = 0; g < 2; ++g)
#pragma unroll
        for (int c = 0; c < 4; ++c)
#pragma unroll
            for (int v = 0; v < 8; ++v) {
                const int row = i0 + ti + g * 16 + v + (hi ? 8 : 0);
                const int col = j0 + tj + c * 16 + l;
                energy[((size_t)(b * CAM_C + row)) * CAM_C + col] = acc[g][c][v];
                if (i0 != j0)
                    energy[((size_t)(b * CAM_C + col)) * CAM_C + row] = acc[g][c][v];
            }
}

// ---------------------------------------------------------------------------
// Kernel 2: attention = softmax(rowmax - energy) == exp(rowmin - e)/sum, per
// row of 512; one wave per row, output as bf16 (A-matrix feed for kernel 3).
// grid = B*C/8 = 1024, block = 256 (8 waves)
// ---------------------------------------------------------------------------
__global__ __launch_bounds__(256) void k2_softmax(
    const float* __restrict__ energy,
    unsigned short* __restrict__ attn)
{
    const int wid  = threadIdx.x >> 5;
    const int lane = threadIdx.x & 31;
    const int row  = blockIdx.x * 8 + wid;

    const float* e = energy + (size_t)row * CAM_C;
    float vals[16];
    float mn = 3.4e38f;
#pragma unroll
    for (int i = 0; i < 16; ++i) {
        vals[i] = e[lane + i * 32];
        mn = fminf(mn, vals[i]);
    }
#pragma unroll
    for (int m = 16; m >= 1; m >>= 1) mn = fminf(mn, __shfl_xor(mn, m, 32));

    float s = 0.0f;
#pragma unroll
    for (int i = 0; i < 16; ++i) {
        vals[i] = __expf(mn - vals[i]);
        s += vals[i];
    }
#pragma unroll
    for (int m = 16; m >= 1; m >>= 1) s += __shfl_xor(s, m, 32);

    const float inv = 1.0f / s;
    unsigned short* a = attn + (size_t)row * CAM_C;
#pragma unroll
    for (int i = 0; i < 16; ++i) a[lane + i * 32] = f2bf(vals[i] * inv);
}

// ---------------------------------------------------------------------------
// Kernel 3: out = x + gamma * softmax_N( attention @ v ).
// One WG (512 threads = 16 waves) per 16 output rows of one batch.
// Attention rows staged in LDS (A); B loaded from transposed bf16 copy.
// Each wave processes 4 n-tiles per iteration (one A gather feeds 4 WMMAs,
// 8 global b128 loads in flight). Two passes: online (max,sum), then emit.
// grid = B * C/16 = 512, block = 512
// ---------------------------------------------------------------------------
__global__ __launch_bounds__(512) void k3_attnv(
    const unsigned short* __restrict__ attn,
    const unsigned short* __restrict__ xbT,
    const float* __restrict__ x,
    const float* __restrict__ gamma,
    float* __restrict__ out)
{
    __shared__ unsigned int aLds[4096];   // 16 rows x 512 bf16 = 16 KB
    __shared__ float sMax[16 * 16];
    __shared__ float sSum[16 * 16];
    __shared__ float fM[16];
    __shared__ float fS[16];

    const int b  = blockIdx.x >> 5;
    const int m0 = (blockIdx.x & 31) << 4;
    const int t  = threadIdx.x;
    const int lane = t & 31;
    const int wid  = t >> 5;
    const bool hi  = lane >= 16;
    const int  l   = lane & 15;

#pragma unroll
    for (int p = 0; p < 2; ++p) {
        const int idx = p * 512 + t;
        const int r   = idx >> 6;
        const int seg = idx & 63;
        ((uint4*)aLds)[idx] =
            *(const uint4*)(attn + ((size_t)(b * CAM_C + m0 + r)) * CAM_C + seg * 8);
    }
    __syncthreads();

    float rmax[8], rsum[8];
#pragma unroll
    for (int v = 0; v < 8; ++v) { rmax[v] = -3.4e38f; rsum[v] = 0.0f; }

    // ---- pass 1: statistics (4 n-tiles per iteration) ----
    for (int it = 0; it < 4; ++it) {
        const int n0 = wid * 256 + it * 64;
        v8f acc[4];
#pragma unroll
        for (int c = 0; c < 4; ++c)
#pragma unroll
            for (int i = 0; i < 8; ++i) acc[c][i] = 0.0f;

#pragma unroll
        for (int kk = 0; kk < CAM_C / 32; ++kk) {
            Frag16 af, bf[4];
            const int ab = l * 256 + kk * 16 + (hi ? 8 : 0);
            af.q[0] = *(const uint4*)&aLds[ab];
            af.q[1] = *(const uint4*)&aLds[ab + 4];
#pragma unroll
            for (int c = 0; c < 4; ++c) {
                const unsigned short* bp =
                    xbT + ((size_t)(b * CAM_N + n0 + c * 16 + l)) * CAM_C + kk * 32 + (hi ? 16 : 0);
                bf[c].q[0] = *(const uint4*)bp;
                bf[c].q[1] = *(const uint4*)(bp + 8);
            }
#pragma unroll
            for (int c = 0; c < 4; ++c)
                acc[c] = __builtin_amdgcn_wmma_f32_16x16x32_bf16(
                    false, af.v, false, bf[c].v, (short)0, acc[c], false, false);
        }
#pragma unroll
        for (int c = 0; c < 4; ++c)
#pragma unroll
            for (int v = 0; v < 8; ++v) {
                float m = acc[c][v];
#pragma unroll
                for (int msk = 8; msk >= 1; msk >>= 1) m = fmaxf(m, __shfl_xor(m, msk, 32));
                const float M = fmaxf(rmax[v], m);
                float e = __expf(acc[c][v] - M);
#pragma unroll
                for (int msk = 8; msk >= 1; msk >>= 1) e += __shfl_xor(e, msk, 32);
                rsum[v] = rsum[v] * __expf(rmax[v] - M) + e;
                rmax[v] = M;
            }
    }
    if (l == 0) {
        const int rbase = hi ? 8 : 0;
#pragma unroll
        for (int v = 0; v < 8; ++v) {
            sMax[wid * 16 + rbase + v] = rmax[v];
            sSum[wid * 16 + rbase + v] = rsum[v];
        }
    }
    __syncthreads();
    if (t < 16) {
        float M = -3.4e38f;
        for (int w = 0; w < 16; ++w) M = fmaxf(M, sMax[w * 16 + t]);
        float S = 0.0f;
        for (int w = 0; w < 16; ++w) S += sSum[w * 16 + t] * __expf(sMax[w * 16 + t] - M);
        fM[t] = M;
        fS[t] = S;
    }
    __syncthreads();

    const float g = gamma[0];

    // ---- pass 2: recompute + emit ----
    for (int it = 0; it < 4; ++it) {
        const int n0 = wid * 256 + it * 64;
        v8f acc[4];
#pragma unroll
        for (int c = 0; c < 4; ++c)
#pragma unroll
            for (int i = 0; i < 8; ++i) acc[c][i] = 0.0f;

#pragma unroll
        for (int kk = 0; kk < CAM_C / 32; ++kk) {
            Frag16 af, bf[4];
            const int ab = l * 256 + kk * 16 + (hi ? 8 : 0);
            af.q[0] = *(const uint4*)&aLds[ab];
            af.q[1] = *(const uint4*)&aLds[ab + 4];
#pragma unroll
            for (int c = 0; c < 4; ++c) {
                const unsigned short* bp =
                    xbT + ((size_t)(b * CAM_N + n0 + c * 16 + l)) * CAM_C + kk * 32 + (hi ? 16 : 0);
                bf[c].q[0] = *(const uint4*)bp;
                bf[c].q[1] = *(const uint4*)(bp + 8);
            }
#pragma unroll
            for (int c = 0; c < 4; ++c)
                acc[c] = __builtin_amdgcn_wmma_f32_16x16x32_bf16(
                    false, af.v, false, bf[c].v, (short)0, acc[c], false, false);
        }
#pragma unroll
        for (int c = 0; c < 4; ++c)
#pragma unroll
            for (int v = 0; v < 8; ++v) {
                const int row = (hi ? 8 : 0) + v;
                const int col = n0 + c * 16 + l;
                const float sm = __expf(acc[c][v] - fM[row]) / fS[row];
                const size_t idx = ((size_t)(b * CAM_C + m0 + row)) * CAM_N + col;
                out[idx] = x[idx] + g * sm;
            }
    }
}

// ---------------------------------------------------------------------------
// Launcher
// ---------------------------------------------------------------------------
extern "C" void kernel_launch(void* const* d_in, const int* in_sizes, int n_in,
                              void* d_out, int out_size, void* d_ws, size_t ws_size,
                              hipStream_t stream)
{
    (void)in_sizes; (void)n_in; (void)out_size; (void)ws_size;

    const float* x     = (const float*)d_in[0];
    const float* gamma = (const float*)d_in[1];
    float*       out   = (float*)d_out;

    char* ws = (char*)d_ws;
    unsigned short* xb     = (unsigned short*)(ws);                          // 64 MB
    unsigned short* xbT    = (unsigned short*)(ws + (size_t)67108864);       // 64 MB
    float*          energy = (float*)         (ws + (size_t)134217728);      // 16 MB
    unsigned short* attn   = (unsigned short*)(ws + (size_t)150994944);      //  8 MB

    k0_convert<<<8192, 256, 0, stream>>>(x, xb, xbT);
    k1_gram   <<< 160, 256, 0, stream>>>(xb, energy);
    k2_softmax<<<1024, 256, 0, stream>>>(energy, attn);
    k3_attnv  <<< 512, 512, 0, stream>>>(attn, xbT, x, gamma, out);
}
